// LSTM_2250562863751
// MI455X (gfx1250) — matrix operand order repaired and checked by reading, not compiled, not run on hardware
//
#include <hip/hip_runtime.h>
#include <hip/hip_bf16.h>
#include <stdint.h>

// ---------------------------------------------------------------------------
// LSTM (2 layers) for MI455X / gfx1250, wave32 + WMMA bf16.
// B=64, S=512, I=H=1024, 4H=4096.
// Register-blocked MTxNT WMMA tiles: (MT+NT)/(MT*NT) KB of fragment traffic
// per v_wmma_f32_16x16x32_bf16.
// ---------------------------------------------------------------------------

#define BATCH 64
#define SEQ   512
#define HDIM  1024
#define KDIM  1024          // I == H == 1024 for both layers
#define GDIM  4096          // 4*H

typedef __bf16 bf16_t;
typedef __attribute__((ext_vector_type(16))) __bf16   v16bf;
typedef __attribute__((ext_vector_type(8)))  float    v8f;
typedef uint32_t v4u __attribute__((ext_vector_type(4)));

union FragU { v4u q[2]; v16bf v; };

__device__ __forceinline__ unsigned short f2bf(float f) {
    union { float f; uint32_t u; } v; v.f = f;
    uint32_t u = v.u;
    u += 0x7FFFu + ((u >> 16) & 1u);      // round-to-nearest-even
    return (unsigned short)(u >> 16);
}

__device__ __forceinline__ float sigmoidf(float x) {
    return 1.0f / (1.0f + expf(-x));
}

// A fragment: 16x32 bf16, row-major source (lda = K).
// ISA layout: lanes 0-15 -> M=lane, K chunks [0..7] and [16..23];
//             lanes 16-31 -> M=lane-16, K chunks [8..15] and [24..31].
__device__ __forceinline__ v16bf load_a_frag(const unsigned short* A, int lda,
                                             int mBase, int kBase, int lane) {
    int m    = mBase + (lane & 15);
    int koff = (lane >> 4) * 8;
    const unsigned short* p = A + (size_t)m * lda + kBase + koff;
    FragU f;
    f.q[0] = *(const v4u*)(p);        // K = koff .. koff+7
    f.q[1] = *(const v4u*)(p + 16);   // K = koff+16 .. koff+23
    return f.v;
}

// B fragment: 32x16 bf16. Weights are pre-packed N-major (Bp[n*K + k]) so each
// lane reads two 8-element K chunks (two 16-byte reads):
// lanes 0-15 -> N=lane, K [0..7] and [16..23]; lanes 16-31 -> N=lane-16, K [8..15] and [24..31].
__device__ __forceinline__ v16bf load_b_frag(const unsigned short* Bp,
                                             int nBase, int kBase, int lane) {
    int n    = nBase + (lane & 15);
    int koff = (lane >> 4) * 8;
    const unsigned short* p = Bp + (size_t)n * KDIM + kBase + koff;
    FragU f;
    f.q[0] = *(const v4u*)(p);        // K = koff .. koff+7
    f.q[1] = *(const v4u*)(p + 16);   // K = koff+16 .. koff+23
    return f.v;
}

// MT 16-row M tiles x NT 16-col N tiles, full K reduction.
// A frags loaded once per k-slice, each reused across NT WMMAs; each B frag
// reused across MT WMMAs.
template <int MT, int NT>
__device__ __forceinline__ void tile_gemm(const unsigned short* A, int lda,
                                          const unsigned short* Bp,
                                          int mBase, int nBase, int lane,
                                          v8f (&acc)[MT][NT]) {
    v8f z;
    #pragma unroll
    for (int e = 0; e < 8; ++e) z[e] = 0.0f;
    #pragma unroll
    for (int m = 0; m < MT; ++m)
        #pragma unroll
        for (int i = 0; i < NT; ++i) acc[m][i] = z;

    for (int k = 0; k < KDIM; k += 32) {
        // Pull the next k-slice of the streaming (weight) operand toward the
        // WGP while this slice's WMMAs execute (global_prefetch_b8).
        if (k + 32 < KDIM) {
            const unsigned short* pf =
                Bp + (size_t)(nBase + (lane & 15)) * KDIM + (k + 32) + (lane >> 4) * 16;
            __builtin_prefetch(pf, 0, 1);
        }
        v16bf a[MT];
        #pragma unroll
        for (int m = 0; m < MT; ++m)
            a[m] = load_a_frag(A, lda, mBase + 16 * m, k, lane);
        #pragma unroll
        for (int i = 0; i < NT; ++i) {
            v16bf b = load_b_frag(Bp, nBase + 16 * i, k, lane);
            #pragma unroll
            for (int m = 0; m < MT; ++m)
                acc[m][i] = __builtin_amdgcn_wmma_f32_16x16x32_bf16(
                    false, a[m], false, b, (short)0, acc[m][i], false, false);
        }
    }
}

// ------------------------- weight / activation prep -------------------------

// W: fp32 (KDIM x GDIM) row-major  ->  Wp: bf16 (GDIM x KDIM)  (transposed)
__global__ void lstm_pack_weight(const float* __restrict__ W,
                                 unsigned short* __restrict__ Wp) {
    int idx = blockIdx.x * blockDim.x + threadIdx.x;
    if (idx >= KDIM * GDIM) return;
    int k = idx / GDIM;
    int n = idx - k * GDIM;
    Wp[(size_t)n * KDIM + k] = f2bf(W[idx]);
}

__global__ void lstm_cvt_bf16(const float* __restrict__ X,
                              unsigned short* __restrict__ Xb, int n) {
    int idx = blockIdx.x * blockDim.x + threadIdx.x;
    if (idx < n) Xb[idx] = f2bf(X[idx]);
}

// ------------------------- GEMM kernels -------------------------

// Xg[m][n] = sum_k A[m][k]*Wp[n][k] + bx[n] + bh[n]     (M = BATCH*SEQ rows)
template <int MT, int NT>
__global__ void lstm_xg_gemm(const unsigned short* __restrict__ A,
                             const unsigned short* __restrict__ Wp,
                             const float* __restrict__ bx,
                             const float* __restrict__ bh,
                             float* __restrict__ Xg) {
    const int wavesPerBlk = blockDim.x >> 5;
    int wave = threadIdx.x >> 5;
    int lane = threadIdx.x & 31;
    int task = blockIdx.x * wavesPerBlk + wave;           // wave-uniform
    const int nGroups = (GDIM / 16) / NT;
    int mG = task / nGroups;
    int nG = task - mG * nGroups;
    if (mG >= ((BATCH * SEQ) / 16) / MT) return;          // uniform exit
    int mBase = mG * (16 * MT);
    int nBase = nG * (16 * NT);

    v8f acc[MT][NT];
    tile_gemm<MT, NT>(A, KDIM, Wp, mBase, nBase, lane, acc);

    #pragma unroll
    for (int m = 0; m < MT; ++m) {
        int rbase = mBase + 16 * m + ((lane >> 4) & 1) * 8;
        #pragma unroll
        for (int i = 0; i < NT; ++i) {
            int col  = nBase + 16 * i + (lane & 15);
            float bias = bx[col] + bh[col];
            #pragma unroll
            for (int r = 0; r < 8; ++r)
                Xg[(size_t)(rbase + r) * GDIM + col] = acc[m][i][r] + bias;
        }
    }
}

// gates[b][n] = sum_k Hbf[b][k]*Whp[n][k] + Xg[(b*SEQ + t)][n]     (M = BATCH)
template <int MT, int NT>
__global__ void lstm_step_gemm(const unsigned short* __restrict__ Hbf,
                               const unsigned short* __restrict__ Whp,
                               const float* __restrict__ Xg, int t,
                               float* __restrict__ gates) {
    const int wavesPerBlk = blockDim.x >> 5;
    int wave = threadIdx.x >> 5;
    int lane = threadIdx.x & 31;
    int task = blockIdx.x * wavesPerBlk + wave;           // wave-uniform
    const int nGroups = (GDIM / 16) / NT;
    int mG = task / nGroups;
    int nG = task - mG * nGroups;
    if (mG >= (BATCH / 16) / MT) return;                  // uniform exit
    int mBase = mG * (16 * MT);
    int nBase = nG * (16 * NT);

    v8f acc[MT][NT];
    tile_gemm<MT, NT>(Hbf, HDIM, Whp, mBase, nBase, lane, acc);

    #pragma unroll
    for (int m = 0; m < MT; ++m) {
        int rbase = mBase + 16 * m + ((lane >> 4) & 1) * 8;
        #pragma unroll
        for (int i = 0; i < NT; ++i) {
            int col = nBase + 16 * i + (lane & 15);
            #pragma unroll
            for (int r = 0; r < 8; ++r) {
                int b = rbase + r;
                gates[(size_t)b * GDIM + col] =
                    acc[m][i][r] + Xg[((size_t)b * SEQ + t) * GDIM + col];
            }
        }
    }
}

// ------------------------- fused LSTM cell -------------------------

// seq_f32 / seq_bf indexed as [(b*SEQ + t)*HDIM + j]; either may be null.
__global__ void lstm_cell(const float* __restrict__ gates,
                          float* __restrict__ h, float* __restrict__ c,
                          unsigned short* __restrict__ hbf,
                          float* __restrict__ seq_f32,
                          unsigned short* __restrict__ seq_bf, int t) {
    int idx = blockIdx.x * blockDim.x + threadIdx.x;      // 0 .. BATCH*HDIM-1
    int b = idx >> 10;
    int j = idx & (HDIM - 1);
    const float* g = gates + (size_t)b * GDIM;
    float ig = g[j];
    float fg = g[HDIM + j];
    float gg = g[2 * HDIM + j];
    float og = g[3 * HDIM + j];

    float cn = sigmoidf(fg) * c[idx] + sigmoidf(ig) * tanhf(gg);
    float hn = sigmoidf(og) * tanhf(cn);

    c[idx]   = cn;
    h[idx]   = hn;
    hbf[idx] = f2bf(hn);
    size_t sidx = ((size_t)b * SEQ + t) * HDIM + j;
    if (seq_f32) seq_f32[sidx] = hn;
    if (seq_bf)  seq_bf[sidx]  = f2bf(hn);
}

__global__ void lstm_copy_state(const float* __restrict__ h,
                                const float* __restrict__ c,
                                float* __restrict__ out_h,
                                float* __restrict__ out_c) {
    int idx = blockIdx.x * blockDim.x + threadIdx.x;
    out_h[idx] = h[idx];
    out_c[idx] = c[idx];
}

// ---------------------------------------------------------------------------
// Host launcher
// ---------------------------------------------------------------------------

extern "C" void kernel_launch(void* const* d_in, const int* in_sizes, int n_in,
                              void* d_out, int out_size, void* d_ws, size_t ws_size,
                              hipStream_t stream) {
    const float* x   = (const float*)d_in[0];
    const float* Wx0 = (const float*)d_in[1];
    const float* bx0 = (const float*)d_in[2];
    const float* Wh0 = (const float*)d_in[3];
    const float* bh0 = (const float*)d_in[4];
    const float* Wx1 = (const float*)d_in[5];
    const float* bx1 = (const float*)d_in[6];
    const float* Wh1 = (const float*)d_in[7];
    const float* bh1 = (const float*)d_in[8];
    float* out = (float*)d_out;

    uint8_t* ws = (uint8_t*)d_ws;
    size_t off = 0;
    const size_t W_BYTES = (size_t)GDIM * KDIM * sizeof(unsigned short); // 8 MB
    unsigned short* wxp0 = (unsigned short*)(ws + off); off += W_BYTES;
    unsigned short* whp0 = (unsigned short*)(ws + off); off += W_BYTES;
    unsigned short* wxp1 = (unsigned short*)(ws + off); off += W_BYTES;
    unsigned short* whp1 = (unsigned short*)(ws + off); off += W_BYTES;
    unsigned short* x_bf    = (unsigned short*)(ws + off);
    off += (size_t)BATCH * SEQ * KDIM * sizeof(unsigned short);          // 64 MB
    unsigned short* hseq_bf = (unsigned short*)(ws + off);
    off += (size_t)BATCH * SEQ * HDIM * sizeof(unsigned short);          // 64 MB
    float* xg = (float*)(ws + off);
    off += (size_t)BATCH * SEQ * GDIM * sizeof(float);                   // 512 MB
    float* gates = (float*)(ws + off);
    off += (size_t)BATCH * GDIM * sizeof(float);                         // 1 MB
    float* h_state = (float*)(ws + off); off += (size_t)BATCH * HDIM * sizeof(float);
    float* c_state = (float*)(ws + off); off += (size_t)BATCH * HDIM * sizeof(float);
    unsigned short* h_bf = (unsigned short*)(ws + off);
    off += (size_t)BATCH * HDIM * sizeof(unsigned short);

    // ---- one-time prep: bf16 weight repack (transposed) + bf16 input ----
    {
        int n = KDIM * GDIM;
        int blks = (n + 255) / 256;
        lstm_pack_weight<<<blks, 256, 0, stream>>>(Wx0, wxp0);
        lstm_pack_weight<<<blks, 256, 0, stream>>>(Wh0, whp0);
        lstm_pack_weight<<<blks, 256, 0, stream>>>(Wx1, wxp1);
        lstm_pack_weight<<<blks, 256, 0, stream>>>(Wh1, whp1);
    }
    {
        int n = BATCH * SEQ * KDIM;
        lstm_cvt_bf16<<<(n + 255) / 256, 256, 0, stream>>>(x, x_bf, n);
    }

    const size_t OUT_SEQ = (size_t)BATCH * SEQ * HDIM;   // 33,554,432 floats
    const int BH = BATCH * HDIM;                          // 65,536

    for (int layer = 0; layer < 2; ++layer) {
        const unsigned short* A   = layer ? hseq_bf : x_bf;
        const unsigned short* wxp = layer ? wxp1 : wxp0;
        const unsigned short* whp = layer ? whp1 : whp0;
        const float* bx = layer ? bx1 : bx0;
        const float* bh = layer ? bh1 : bh0;

        // xg = A @ Wx + bx + bh : M=32768, N=4096, K=1024
        // MT=4,NT=2 -> tasks = (2048/4)*(256/2) = 65536 waves / 8 = 8192 blocks
        lstm_xg_gemm<4, 2><<<8192, 256, 0, stream>>>(A, wxp, bx, bh, xg);

        hipMemsetAsync(h_state, 0, (size_t)BH * sizeof(float), stream);
        hipMemsetAsync(c_state, 0, (size_t)BH * sizeof(float), stream);
        hipMemsetAsync(h_bf,    0, (size_t)BH * sizeof(unsigned short), stream);

        float*          seq_f  = layer ? out : nullptr;
        unsigned short* seq_bf = layer ? nullptr : hseq_bf;

        for (int t = 0; t < SEQ; ++t) {
            // gates = xg[:,t,:] + h @ Wh : M=64, N=4096, K=1024
            // MT=4 covers all of M -> each Wh column group loaded exactly once
            // per step. tasks = 1*(256/2) = 128 waves / 8 = 16 blocks.
            lstm_step_gemm<4, 2><<<16, 256, 0, stream>>>(h_bf, whp, xg, t, gates);
            lstm_cell<<<BH / 256, 256, 0, stream>>>(gates, h_state, c_state,
                                                    h_bf, seq_f, seq_bf, t);
        }

        // final (hT, cT) for this layer -> stacked outputs
        lstm_copy_state<<<BH / 256, 256, 0, stream>>>(
            h_state, c_state,
            out + OUT_SEQ + (size_t)layer * BH,
            out + OUT_SEQ + 2 * (size_t)BH + (size_t)layer * BH);
    }
}
